// GoalInferenceNetwork_56607668961920
// MI455X (gfx1250) — compile-verified, matrix operand order checked by reading, not run on hardware
//
#include <hip/hip_runtime.h>
#include <hip/hip_bf16.h>

// Problem constants
#define BATCH 32768
#define TOBS  20
#define NAG   4
#define DDIM  8
#define HDIM  64

typedef __attribute__((ext_vector_type(16))) _Float16 v16h;
typedef __attribute__((ext_vector_type(8)))  _Float16 v8h;
typedef __attribute__((ext_vector_type(8)))  float    v8f;

// ---------------------------------------------------------------------------
// CDNA5 16x16x32 f16 fragment layout (CDNA5 ISA):
//  A (16x32): lane L: M=L&15; element j covers K = h8 + j (j<8) and
//             K = h8 + 16 + (j-8) (j>=8), where h8 = (L>>4)*8.
//    -> per lane: two contiguous 8-half runs  ==> 2x 16B vector loads.
//  B (32x16): lane L: N=L&15; same K mapping  ==> store weights transposed
//             [col][k] so B-fragments are also two contiguous 8-half runs.
//  C/D (16x16 f32): lane L: N=L&15; vgpr r: M = r + ((L>>4)<<3).
// ---------------------------------------------------------------------------

__device__ __forceinline__ v8f wmma32(v16h a, v16h b, v8f c) {
    return __builtin_amdgcn_wmma_f32_16x16x32_f16(false, a, false, b, (short)0, c, false, false);
}

__device__ __forceinline__ v8f zero8() {
    v8f z = {0.f, 0.f, 0.f, 0.f, 0.f, 0.f, 0.f, 0.f};
    return z;
}

__device__ __forceinline__ v16h pack2(v8h lo, v8h hi) {
    v16h a;
#pragma unroll
    for (int j = 0; j < 8; ++j) { a[j] = lo[j]; a[j + 8] = hi[j]; }
    return a;
}

// A-fragment from f16 row-major [16][ldk]; base pre-offset by K-base.
__device__ __forceinline__ v16h load_a16(const _Float16* base, int ldk, int lane) {
    const _Float16* p = base + (lane & 15) * ldk + ((lane >> 4) << 3);
    return pack2(*(const v8h*)p, *(const v8h*)(p + 16));
}

// B-fragment from TRANSPOSED f16 weights in LDS: wt[col][k].
__device__ __forceinline__ v16h load_bT(const _Float16* wt, int ldk, int c0, int lane) {
    const _Float16* p = wt + (c0 + (lane & 15)) * ldk + ((lane >> 4) << 3);
    return pack2(*(const v8h*)p, *(const v8h*)(p + 16));
}

// B-fragment straight from global f32 row-major [K][ld] (scalar column loads)
__device__ __forceinline__ v16h load_b_gf32(const float* w, int ld, int c0, int lane) {
    int n = c0 + (lane & 15);
    int h8 = (lane >> 4) << 3;
    v16h b;
#pragma unroll
    for (int j = 0; j < 8; ++j) {
        b[j]     = (_Float16)w[(h8 + j) * ld + n];
        b[j + 8] = (_Float16)w[(h8 + 16 + j) * ld + n];
    }
    return b;
}

// gfx1250 hardware tanh (TRANS op). Builtin preferred; asm fallback keeps the
// ISA-mandated 1-op spacing after a TRANS op via a trailing v_nop.
__device__ __forceinline__ float tanh_hw(float x) {
#if __has_builtin(__builtin_amdgcn_tanhf)
    return __builtin_amdgcn_tanhf(x);
#else
    float r;
    asm volatile("v_tanh_f32 %0, %1\n\tv_nop" : "=v"(r) : "v"(x));
    return r;
#endif
}

// sigmoid(x) = 0.5*tanh(x/2) + 0.5  (one TRANS + one FMA)
__device__ __forceinline__ float sigmoid_hw(float x) {
    return __builtin_fmaf(0.5f, tanh_hw(0.5f * x), 0.5f);
}

// ---------------------------------------------------------------------------
// Kernel 1: per-agent GRU over T=20 steps.
// grid = (BATCH/64, NAG), block = 128 (4 waves). Wave w handles 16 batch rows
// of agent blockIdx.y. Emits feat[B][256] (f16, row-major) into workspace.
// No occupancy cap: let the compiler pin the 36 loop-invariant weight
// fragments in VGPRs (~2 waves/SIMD, dense WMMA chain, zero spills) —
// capping to 4 waves/SIMD made it spill fragments to scratch (round 3).
// ---------------------------------------------------------------------------
__global__ __launch_bounds__(128)
void gru_kernel(const float* __restrict__ x,   // [B][T*N*D]
                const float* __restrict__ Wi,  // [N][D][3H]
                const float* __restrict__ Wh,  // [N][H][3H]
                const float* __restrict__ bh,  // [N][3H]
                _Float16* __restrict__ feat)   // [B][N*H]
{
    __shared__ alignas(16) _Float16 s_wiT[3 * HDIM * 32];   // transposed, K padded to 32 (k>=8 zero)
    __shared__ alignas(16) _Float16 s_whT[3 * HDIM * HDIM]; // transposed [192][64]
    __shared__ alignas(16) float    s_bh[3 * HDIM];
    __shared__ alignas(16) _Float16 s_x[64 * TOBS * DDIM];  // [64 rows][20 t][8 d]
    __shared__ alignas(16) _Float16 s_hA[4 * 16 * HDIM];    // per-wave h, row-major [16][64]

    const int tid  = threadIdx.x;
    const int lane = tid & 31;
    const int w    = tid >> 5;
    const int n    = blockIdx.y;           // agent
    const int r0   = blockIdx.x * 64;      // first batch row of block

    // ---- Phase 0: stage weights (transposed f16) + x slices ----
    for (int i = tid; i < 3 * HDIM * 32; i += 128) {
        int c = i >> 5, k = i & 31;
        _Float16 v = (_Float16)Wi[n * DDIM * 3 * HDIM + (k & 7) * (3 * HDIM) + c];
        s_wiT[i] = (k < DDIM) ? v : (_Float16)0.0f;   // value select, no branchy load
    }
    for (int i = tid; i < 3 * HDIM * HDIM; i += 128) {
        int c = i >> 6, k = i & 63;
        s_whT[i] = (_Float16)Wh[n * HDIM * 3 * HDIM + k * (3 * HDIM) + c];
    }
    for (int i = tid; i < 3 * HDIM; i += 128)
        s_bh[i] = bh[n * 3 * HDIM + i];
    for (int i = tid; i < 64 * TOBS * DDIM; i += 128) {
        int rr = i / (TOBS * DDIM);
        int rem = i - rr * (TOBS * DDIM);
        int t = rem >> 3, d = rem & 7;
        s_x[i] = (_Float16)x[(size_t)(r0 + rr) * (TOBS * NAG * DDIM) + (t * NAG + n) * DDIM + d];
    }
    for (int i = tid; i < 4 * 16 * HDIM; i += 128) s_hA[i] = (_Float16)0.0f;
    __syncthreads();

    _Float16* myhA = s_hA + w * 16 * HDIM;                 // [16][64]
    const _Float16* myx = s_x + (w * 16) * (TOBS * DDIM);

    v8f h[4];
#pragma unroll
    for (int j = 0; j < 4; ++j) h[j] = zero8();

    const int colL = lane & 15;
    const int rowH = (lane >> 4) << 3;
    const bool lo16 = (lane < 16);

    // ---- Phase 1: GRU scan ----
    for (int t = 0; t < TOBS; ++t) {
        // x A-fragment: one 16B LDS load + value select (lanes>=16 and k>=8 are zero)
        v16h xa;
        {
            v8h v = *(const v8h*)(myx + (lane & 15) * (TOBS * DDIM) + t * DDIM);
#pragma unroll
            for (int j = 0; j < 8; ++j) {
                xa[j]     = lo16 ? v[j] : (_Float16)0.0f;
                xa[j + 8] = (_Float16)0.0f;
            }
        }
        // h A-fragments (K slices 0..31, 32..63): 2x 16B LDS loads each
        v16h ha0 = load_a16(myhA, HDIM, lane);
        v16h ha1 = load_a16(myhA + 32, HDIM, lane);

        v8f rg[4], zg[4];
#pragma unroll
        for (int j = 0; j < 4; ++j) {                         // r gate, cols 16j
            int c = 16 * j;
            v8f acc = zero8();
            acc = wmma32(ha0, load_bT(s_whT, HDIM, c, lane), acc);
            acc = wmma32(ha1, load_bT(s_whT + 32, HDIM, c, lane), acc);
            acc = wmma32(xa, load_bT(s_wiT, 32, c, lane), acc);
            float bb = s_bh[c + colL];
#pragma unroll
            for (int r = 0; r < 8; ++r) rg[j][r] = sigmoid_hw(acc[r] + bb);
        }
#pragma unroll
        for (int j = 0; j < 4; ++j) {                         // z gate, cols 64+16j
            int c = 64 + 16 * j;
            v8f acc = zero8();
            acc = wmma32(ha0, load_bT(s_whT, HDIM, c, lane), acc);
            acc = wmma32(ha1, load_bT(s_whT + 32, HDIM, c, lane), acc);
            acc = wmma32(xa, load_bT(s_wiT, 32, c, lane), acc);
            float bb = s_bh[c + colL];
#pragma unroll
            for (int r = 0; r < 8; ++r) zg[j][r] = sigmoid_hw(acc[r] + bb);
        }
#pragma unroll
        for (int j = 0; j < 4; ++j) {                         // n gate, cols 128+16j
            int c = 128 + 16 * j;
            v8f hacc = zero8();                               // hidden part (bias, scaled by r)
            hacc = wmma32(ha0, load_bT(s_whT, HDIM, c, lane), hacc);
            hacc = wmma32(ha1, load_bT(s_whT + 32, HDIM, c, lane), hacc);
            v8f iacc = zero8();                               // input part
            iacc = wmma32(xa, load_bT(s_wiT, 32, c, lane), iacc);
            float bb = s_bh[c + colL];
#pragma unroll
            for (int r = 0; r < 8; ++r) {
                float nn = tanh_hw(__builtin_fmaf(rg[j][r], hacc[r] + bb, iacc[r]));
                h[j][r] = __builtin_fmaf(zg[j][r], h[j][r] - nn, nn);  // (1-z)n + z h
            }
        }
        // write h (C layout) back to LDS row-major for next step's A-fragments
#pragma unroll
        for (int j = 0; j < 4; ++j)
#pragma unroll
            for (int r = 0; r < 8; ++r)
                myhA[(r + rowH) * HDIM + 16 * j + colL] = (_Float16)h[j][r];
    }

    // ---- Phase 2: emit feat with 16B vector global stores (myhA is final h) ----
#pragma unroll
    for (int c8 = 0; c8 < 4; ++c8) {                 // 128 chunks of 8 halves / 32 lanes
        int chunk = lane + 32 * c8;
        int row = chunk >> 3;
        int koff = (chunk & 7) * 8;
        v8h v = *(const v8h*)(myhA + row * HDIM + koff);
        *(v8h*)(feat + (size_t)(r0 + w * 16 + row) * (NAG * HDIM) + n * HDIM + koff) = v;
    }
}

// ---------------------------------------------------------------------------
// Kernel 2: MLP head 256 -> 256 -> 128 -> 8 with WMMA.
// grid = BATCH/16, block = 128 (4 waves per 16-row tile).
// ---------------------------------------------------------------------------
__global__ __launch_bounds__(128)
void mlp_kernel(const _Float16* __restrict__ feat,  // [B][256]
                const float* __restrict__ W0, const float* __restrict__ b0,
                const float* __restrict__ W1, const float* __restrict__ b1,
                const float* __restrict__ Wout, const float* __restrict__ bout,
                float* __restrict__ out)            // [B][8]
{
    __shared__ alignas(16) _Float16 s_feat[16 * 256];
    __shared__ alignas(16) _Float16 s_h1[16 * 256];
    __shared__ alignas(16) _Float16 s_h2[16 * 128];

    const int tid  = threadIdx.x;
    const int lane = tid & 31;
    const int w    = tid >> 5;
    const int r0   = blockIdx.x * 16;
    const int colL = lane & 15;
    const int rowH = (lane >> 4) << 3;

    // stage feat tile with 16B vector loads (512 chunks / 128 threads)
    for (int c8 = tid; c8 < 16 * 256 / 8; c8 += 128) {
        int row = c8 >> 5;
        int koff = (c8 & 31) * 8;
        *(v8h*)(s_feat + row * 256 + koff) =
            *(const v8h*)(feat + (size_t)(r0 + row) * 256 + koff);
    }
    // warm L2/WGP$ for the next layer's weights (global_prefetch_b8)
    __builtin_prefetch(W1 + tid * 256, 0, 1);
    __syncthreads();

    // ---- Layer 0: [16,256] x [256,256] + b0, relu ----
#pragma unroll
    for (int jt = 0; jt < 4; ++jt) {
        int c0 = (w * 4 + jt) * 16;
        v8f acc = zero8();
#pragma unroll
        for (int kk = 0; kk < 8; ++kk) {
            v16h a = load_a16(s_feat + 32 * kk, 256, lane);
            v16h b = load_b_gf32(W0 + (size_t)(32 * kk) * 256, 256, c0, lane);
            acc = wmma32(a, b, acc);
        }
        float bb = b0[c0 + colL];
#pragma unroll
        for (int r = 0; r < 8; ++r) {
            float v = acc[r] + bb;
            v = v > 0.0f ? v : 0.0f;
            s_h1[(r + rowH) * 256 + c0 + colL] = (_Float16)v;
        }
    }
    __syncthreads();

    // ---- Layer 1: [16,256] x [256,128] + b1, relu ----
#pragma unroll
    for (int jt = 0; jt < 2; ++jt) {
        int c0 = (w * 2 + jt) * 16;
        v8f acc = zero8();
#pragma unroll
        for (int kk = 0; kk < 8; ++kk) {
            v16h a = load_a16(s_h1 + 32 * kk, 256, lane);
            v16h b = load_b_gf32(W1 + (size_t)(32 * kk) * 128, 128, c0, lane);
            acc = wmma32(a, b, acc);
        }
        float bb = b1[c0 + colL];
#pragma unroll
        for (int r = 0; r < 8; ++r) {
            float v = acc[r] + bb;
            v = v > 0.0f ? v : 0.0f;
            s_h2[(r + rowH) * 128 + c0 + colL] = (_Float16)v;
        }
    }
    __syncthreads();

    // ---- Layer 2: [16,128] x [128,8] + bout (one wave, cols padded to 16) ----
    if (w == 0) {
        int cw = colL & 7;                 // clamped column -> unconditional loads
        bool valid = (colL < 8);
        int h8 = rowH;                     // (lane>>4)*8
        v8f acc = zero8();
#pragma unroll
        for (int kk = 0; kk < 4; ++kk) {
            v16h b;
#pragma unroll
            for (int j = 0; j < 8; ++j) {
                _Float16 v0 = (_Float16)Wout[(32 * kk + h8 + j) * 8 + cw];
                _Float16 v1 = (_Float16)Wout[(32 * kk + h8 + 16 + j) * 8 + cw];
                b[j]     = valid ? v0 : (_Float16)0.0f;
                b[j + 8] = valid ? v1 : (_Float16)0.0f;
            }
            v16h a = load_a16(s_h2 + 32 * kk, 128, lane);
            acc = wmma32(a, b, acc);
        }
        if (valid) {
            float bb = bout[colL];
#pragma unroll
            for (int r = 0; r < 8; ++r)
                out[(size_t)(r0 + r + rowH) * 8 + colL] = acc[r] + bb;
        }
    }
}

// ---------------------------------------------------------------------------
extern "C" void kernel_launch(void* const* d_in, const int* in_sizes, int n_in,
                              void* d_out, int out_size, void* d_ws, size_t ws_size,
                              hipStream_t stream) {
    const float* x    = (const float*)d_in[0];
    const float* Wi   = (const float*)d_in[1];
    const float* Wh   = (const float*)d_in[2];
    const float* bh   = (const float*)d_in[3];
    const float* W0   = (const float*)d_in[4];
    const float* b0   = (const float*)d_in[5];
    const float* W1   = (const float*)d_in[6];
    const float* b1   = (const float*)d_in[7];
    const float* Wout = (const float*)d_in[8];
    const float* bout = (const float*)d_in[9];
    float* out = (float*)d_out;

    _Float16* feat = (_Float16*)d_ws;   // BATCH*256 f16 = 16 MB scratch

    dim3 g1(BATCH / 64, NAG);
    gru_kernel<<<g1, 128, 0, stream>>>(x, Wi, Wh, bh, feat);
    mlp_kernel<<<BATCH / 16, 128, 0, stream>>>(feat, W0, b0, W1, b1, Wout, bout, out);
}
